// Attention_15985868275953
// MI455X (gfx1250) — compile-verified, matrix operand order checked by reading, not run on hardware
//
#include <hip/hip_runtime.h>
#include <hip/hip_bf16.h>
#include <cstdint>
#include <cstddef>

#define DIM_   576
#define NHEADS 18
#define KD     32
#define VD     128
#define NTOK   196
#define BATCH  256
#define TOK    (BATCH * NTOK)   // 50176
#define HOUT   3456             // 18 * (32+32+128)
#define DHDIM  2304             // 18 * 128
#define QKSCALE 0.17677669529663689f
#define VSTRIDE 224             // padded token stride for vT (448B, 16B aligned)

typedef _Float16 v8h  __attribute__((ext_vector_type(8)));
typedef _Float16 v16h __attribute__((ext_vector_type(16)));
typedef float    v8f  __attribute__((ext_vector_type(8)));

__device__ __forceinline__ v8f wmma_f16(v16h a, v16h b, v8f c) {
    return __builtin_amdgcn_wmma_f32_16x16x32_f16(false, a, false, b, (short)0, c, false, false);
}

// Row-major fragment: row pointer p (32+ halves of K), lane picks kbase / kbase+16.
__device__ __forceinline__ v16h frag16(const _Float16* p, int kbase) {
    v8h lo = *(const v8h*)(p + kbase);
    v8h hi = *(const v8h*)(p + kbase + 16);
    return __builtin_shufflevector(lo, hi, 0,1,2,3,4,5,6,7,8,9,10,11,12,13,14,15);
}
__device__ __forceinline__ v16h frag16g(const _Float16* p, int kbase, bool valid) {
    v8h lo = {}, hi = {};
    if (valid) {
        lo = *(const v8h*)(p + kbase);
        hi = *(const v8h*)(p + kbase + 16);
    }
    return __builtin_shufflevector(lo, hi, 0,1,2,3,4,5,6,7,8,9,10,11,12,13,14,15);
}

// CDNA5 async global->LDS DMA (per-lane 16B), tracked by ASYNCcnt.
__device__ __forceinline__ void async_ld16(uint32_t ldsAddr, uint64_t gaddr) {
    asm volatile("global_load_async_to_lds_b128 %0, %1, off"
                 :: "v"(ldsAddr), "v"(gaddr) : "memory");
}
__device__ __forceinline__ void wait_async0() {
    asm volatile("s_wait_asynccnt 0x0" ::: "memory");
}

// ---------------- prep kernels ----------------
__global__ void k_prep_wqkv(const float* __restrict__ w, _Float16* __restrict__ wT) {
    int i = blockIdx.x * 256 + threadIdx.x;              // over HOUT*DIM_
    if (i < HOUT * DIM_) {
        int c = i / DIM_, d = i % DIM_;
        wT[i] = (_Float16)w[d * HOUT + c];
    }
}
__global__ void k_prep_wproj(const float* __restrict__ w, _Float16* __restrict__ wT) {
    int i = blockIdx.x * 256 + threadIdx.x;              // over DIM_*DHDIM
    if (i < DIM_ * DHDIM) {
        int e = i / DHDIM, d = i % DHDIM;
        wT[i] = (_Float16)w[d * DIM_ + e];
    }
}
__global__ void k_prep_bias(const float* __restrict__ ab, const int* __restrict__ idx,
                            float* __restrict__ bm, int offset_size) {
    int i = blockIdx.x * 256 + threadIdx.x;              // over NHEADS*NTOK*NTOK
    if (i < NHEADS * NTOK * NTOK) {
        int h = i / (NTOK * NTOK), nm = i % (NTOK * NTOK);
        bm[i] = ab[h * offset_size + idx[nm]];
    }
}

// ---------------- layernorm -> f16 ----------------
__global__ void k_ln(const float* __restrict__ x, const float* __restrict__ w,
                     const float* __restrict__ b, _Float16* __restrict__ xn) {
    int token = blockIdx.x * 8 + (threadIdx.x >> 5);
    int lane  = threadIdx.x & 31;
    const float* xr = x + (size_t)token * DIM_;
    float v[18];
    float s = 0.f;
#pragma unroll
    for (int i = 0; i < 18; ++i) { v[i] = xr[lane + i * 32]; s += v[i]; }
#pragma unroll
    for (int d = 16; d >= 1; d >>= 1) s += __shfl_xor(s, d);
    float mu = s * (1.0f / DIM_);
    float var = 0.f;
#pragma unroll
    for (int i = 0; i < 18; ++i) { float d = v[i] - mu; var += d * d; }
#pragma unroll
    for (int d = 16; d >= 1; d >>= 1) var += __shfl_xor(var, d);
    float rs = rsqrtf(var * (1.0f / DIM_) + 1e-5f);
    _Float16* o = xn + (size_t)token * DIM_;
#pragma unroll
    for (int i = 0; i < 18; ++i) {
        int c = lane + i * 32;
        o[c] = (_Float16)((v[i] - mu) * rs * w[c] + b[c]);
    }
}

// ---------------- QKV GEMM: xn(50176x576) * wqkvT(3456x576)^T ----------------
// Double-buffered LDS, tiles staged with async global->LDS DMA overlapping WMMA.
// Epilogue scatters into q(b,h,196,32)*scale, k(b,h,196,32), vT(b,h,128,VSTRIDE).
__global__ void k_gemm_qkv(const _Float16* __restrict__ A, const _Float16* __restrict__ Bt,
                           const float* __restrict__ bias,
                           _Float16* __restrict__ qO, _Float16* __restrict__ kO,
                           _Float16* __restrict__ vTO) {
    __shared__ __align__(16) _Float16 sA[2][128 * 32];
    __shared__ __align__(16) _Float16 sB[2][128 * 32];
    const int bm = blockIdx.y, bn = blockIdx.x;
    const int t = threadIdx.x, wave = t >> 5, lane = t & 31;
    const int waveM = wave & 3, waveN = wave >> 2;
    const int ln15 = lane & 15, kbase = (lane < 16) ? 0 : 8;

    auto issue = [&](int kc, int b) {
#pragma unroll
        for (int idx = t; idx < 512; idx += 256) {
            int row = idx >> 2, seg = idx & 3;
            async_ld16((uint32_t)(uintptr_t)&sA[b][row * 32 + seg * 8],
                       (uint64_t)(uintptr_t)&A[(size_t)(bm * 128 + row) * DIM_ + kc + seg * 8]);
            async_ld16((uint32_t)(uintptr_t)&sB[b][row * 32 + seg * 8],
                       (uint64_t)(uintptr_t)&Bt[(size_t)(bn * 128 + row) * DIM_ + kc + seg * 8]);
        }
    };

    v8f acc[2][4];
#pragma unroll
    for (int i = 0; i < 2; ++i)
#pragma unroll
        for (int j = 0; j < 4; ++j) acc[i][j] = (v8f){};

    issue(0, 0);
    wait_async0();
    __syncthreads();
    int buf = 0;
    for (int kc = 0; kc < DIM_; kc += 32) {
        if (kc + 32 < DIM_) issue(kc + 32, buf ^ 1);   // DMA next chunk during WMMA
        v16h af[2], bf[4];
#pragma unroll
        for (int mt = 0; mt < 2; ++mt)
            af[mt] = frag16(&sA[buf][(waveM * 32 + mt * 16 + ln15) * 32], kbase);
#pragma unroll
        for (int nt = 0; nt < 4; ++nt)
            bf[nt] = frag16(&sB[buf][(waveN * 64 + nt * 16 + ln15) * 32], kbase);
#pragma unroll
        for (int mt = 0; mt < 2; ++mt)
#pragma unroll
            for (int nt = 0; nt < 4; ++nt)
                acc[mt][nt] = wmma_f16(af[mt], bf[nt], acc[mt][nt]);
        wait_async0();
        __syncthreads();
        buf ^= 1;
    }
    // epilogue scatter
#pragma unroll
    for (int mt = 0; mt < 2; ++mt)
#pragma unroll
        for (int nt = 0; nt < 4; ++nt)
#pragma unroll
            for (int r = 0; r < 8; ++r) {
                int m = r + ((lane < 16) ? 0 : 8);
                int row = bm * 128 + waveM * 32 + mt * 16 + m;
                int col = bn * 128 + waveN * 64 + nt * 16 + ln15;
                float val = acc[mt][nt][r] + bias[col];
                int head = col / 192, c = col % 192;
                int bi = row / NTOK, tok = row % NTOK;
                size_t base = (size_t)(bi * NHEADS + head);
                if (c < 32)
                    qO[(base * NTOK + tok) * KD + c] = (_Float16)(val * QKSCALE);
                else if (c < 64)
                    kO[(base * NTOK + tok) * KD + (c - 32)] = (_Float16)val;
                else
                    vTO[(base * VD + (c - 64)) * VSTRIDE + tok] = (_Float16)val;
            }
}

// ---------------- attention: one block per (b,h) ----------------
__global__ void k_attn(const _Float16* __restrict__ qG, const _Float16* __restrict__ kG,
                       const _Float16* __restrict__ vTG, const float* __restrict__ biasM,
                       _Float16* __restrict__ outG) {
    __shared__ __align__(16) _Float16 sP[8 * 16 * 224];   // per-wave P staging, 57KB
    const int bh = blockIdx.x;
    const int h = bh % NHEADS;
    const int t = threadIdx.x, wave = t >> 5, lane = t & 31;
    const int ln15 = lane & 15, kbase = (lane < 16) ? 0 : 8;
    const _Float16* q  = qG  + (size_t)bh * NTOK * KD;
    const _Float16* k  = kG  + (size_t)bh * NTOK * KD;
    const _Float16* vT = vTG + (size_t)bh * VD * VSTRIDE;
    const float* bias  = biasM + (size_t)h * NTOK * NTOK;
    _Float16* outB = outG + (size_t)(bh / NHEADS) * NTOK * DHDIM + h * VD;

    for (int i = t; i < 8 * 16 * 224 / 2; i += 256) ((int*)sP)[i] = 0;
    __syncthreads();

    _Float16* myP = sP + wave * 16 * 224;

    for (int it = wave; it < 13; it += 8) {
        int qrow = it * 16 + ln15;
        v16h aq = frag16g(q + qrow * KD, kbase, qrow < NTOK);

        float sreg[13][8];
#pragma unroll
        for (int j = 0; j < 13; ++j) {
            int kt = j * 16 + ln15;
            v16h bk = frag16g(k + kt * KD, kbase, kt < NTOK);
            v8f c = (v8f){};
            c = wmma_f16(aq, bk, c);
#pragma unroll
            for (int r = 0; r < 8; ++r) {
                int m = r + ((lane < 16) ? 0 : 8);
                int row = it * 16 + m, col = j * 16 + ln15;
                float sv = -1e30f;
                if (row < NTOK && col < NTOK) sv = c[r] + bias[row * NTOK + col];
                sreg[j][r] = sv;
            }
        }
        // row softmax (rows live in 16-lane halves; columns across lanes 0..15)
        float rinv[8];
#pragma unroll
        for (int r = 0; r < 8; ++r) {
            float mx = sreg[0][r];
#pragma unroll
            for (int j = 1; j < 13; ++j) mx = fmaxf(mx, sreg[j][r]);
#pragma unroll
            for (int d = 8; d >= 1; d >>= 1) mx = fmaxf(mx, __shfl_xor(mx, d));
            float s = 0.f;
#pragma unroll
            for (int j = 0; j < 13; ++j) { float e = __expf(sreg[j][r] - mx); sreg[j][r] = e; s += e; }
#pragma unroll
            for (int d = 8; d >= 1; d >>= 1) s += __shfl_xor(s, d);
            rinv[r] = 1.0f / s;
        }
        // stage P (f16) for A-layout re-read; zero beyond 196 cols
#pragma unroll
        for (int j = 0; j < 13; ++j)
#pragma unroll
            for (int r = 0; r < 8; ++r) {
                int m = r + ((lane < 16) ? 0 : 8);
                int col = j * 16 + ln15;
                _Float16 pv = (_Float16)0.f;
                if (col < NTOK) pv = (_Float16)(sreg[j][r] * rinv[r]);
                myP[m * 224 + col] = pv;
            }
        // out = P(16x224) * V(224xVD); V frags direct from global vT (L0/L2 cached)
        v8f oacc[8];
#pragma unroll
        for (int tt = 0; tt < 8; ++tt) oacc[tt] = (v8f){};
#pragma unroll
        for (int c = 0; c < 7; ++c) {
            v16h ap = frag16(&myP[ln15 * 224 + c * 32], kbase);
#pragma unroll
            for (int tt = 0; tt < 8; ++tt) {
                int dim = tt * 16 + ln15;
                v16h bv = frag16(vT + (size_t)dim * VSTRIDE + c * 32, kbase);
                oacc[tt] = wmma_f16(ap, bv, oacc[tt]);
            }
        }
#pragma unroll
        for (int tt = 0; tt < 8; ++tt)
#pragma unroll
            for (int r = 0; r < 8; ++r) {
                int m = r + ((lane < 16) ? 0 : 8);
                int row = it * 16 + m;
                if (row < NTOK)
                    outB[(size_t)row * DHDIM + tt * 16 + ln15] = (_Float16)oacc[tt][r];
            }
    }
}

// ---------------- proj GEMM: attnout(50176x2304) * wprojT(576x2304)^T + b ----------------
__global__ void k_gemm_proj(const _Float16* __restrict__ A, const _Float16* __restrict__ Bt,
                            const float* __restrict__ bias, float* __restrict__ out) {
    __shared__ __align__(16) _Float16 sA[2][128 * 32];
    __shared__ __align__(16) _Float16 sB[2][128 * 32];
    const int bm = blockIdx.y, bn = blockIdx.x;
    const int t = threadIdx.x, wave = t >> 5, lane = t & 31;
    const int waveM = wave & 3, waveN = wave >> 2;
    const int ln15 = lane & 15, kbase = (lane < 16) ? 0 : 8;

    // B rows >= DIM_ never get DMA'd: zero them once in both buffers.
#pragma unroll
    for (int idx = t; idx < 512; idx += 256) {
        int row = idx >> 2, seg = idx & 3;
        if (bn * 128 + row >= DIM_) {
            *(v8h*)&sB[0][row * 32 + seg * 8] = (v8h){};
            *(v8h*)&sB[1][row * 32 + seg * 8] = (v8h){};
        }
    }

    auto issue = [&](int kc, int b) {
#pragma unroll
        for (int idx = t; idx < 512; idx += 256) {
            int row = idx >> 2, seg = idx & 3;
            async_ld16((uint32_t)(uintptr_t)&sA[b][row * 32 + seg * 8],
                       (uint64_t)(uintptr_t)&A[(size_t)(bm * 128 + row) * DHDIM + kc + seg * 8]);
            int n = bn * 128 + row;
            if (n < DIM_)
                async_ld16((uint32_t)(uintptr_t)&sB[b][row * 32 + seg * 8],
                           (uint64_t)(uintptr_t)&Bt[(size_t)n * DHDIM + kc + seg * 8]);
        }
    };

    v8f acc[2][4];
#pragma unroll
    for (int i = 0; i < 2; ++i)
#pragma unroll
        for (int j = 0; j < 4; ++j) acc[i][j] = (v8f){};

    issue(0, 0);
    wait_async0();
    __syncthreads();
    int buf = 0;
    for (int kc = 0; kc < DHDIM; kc += 32) {
        if (kc + 32 < DHDIM) issue(kc + 32, buf ^ 1);
        v16h af[2], bf[4];
#pragma unroll
        for (int mt = 0; mt < 2; ++mt)
            af[mt] = frag16(&sA[buf][(waveM * 32 + mt * 16 + ln15) * 32], kbase);
#pragma unroll
        for (int nt = 0; nt < 4; ++nt)
            bf[nt] = frag16(&sB[buf][(waveN * 64 + nt * 16 + ln15) * 32], kbase);
#pragma unroll
        for (int mt = 0; mt < 2; ++mt)
#pragma unroll
            for (int nt = 0; nt < 4; ++nt)
                acc[mt][nt] = wmma_f16(af[mt], bf[nt], acc[mt][nt]);
        wait_async0();
        __syncthreads();
        buf ^= 1;
    }
#pragma unroll
    for (int mt = 0; mt < 2; ++mt)
#pragma unroll
        for (int nt = 0; nt < 4; ++nt)
#pragma unroll
            for (int r = 0; r < 8; ++r) {
                int m = r + ((lane < 16) ? 0 : 8);
                int row = bm * 128 + waveM * 32 + mt * 16 + m;
                int col = bn * 128 + waveN * 64 + nt * 16 + ln15;
                if (col < DIM_)
                    out[(size_t)row * DIM_ + col] = acc[mt][nt][r] + bias[col];
            }
}

extern "C" void kernel_launch(void* const* d_in, const int* in_sizes, int n_in,
                              void* d_out, int out_size, void* d_ws, size_t ws_size,
                              hipStream_t stream) {
    const float* x       = (const float*)d_in[0];
    const float* ln_w    = (const float*)d_in[1];
    const float* ln_b    = (const float*)d_in[2];
    const float* w_qkv   = (const float*)d_in[3];
    const float* b_qkv   = (const float*)d_in[4];
    const float* w_proj  = (const float*)d_in[5];
    const float* b_proj  = (const float*)d_in[6];
    const float* attn_b  = (const float*)d_in[7];
    const int*   bidx    = (const int*)d_in[8];
    float* out = (float*)d_out;
    int offset_size = in_sizes[7] / NHEADS;

    char* ws = (char*)d_ws;
    size_t off = 0;
    _Float16* xn      = (_Float16*)(ws + off); off += (size_t)TOK * DIM_ * 2;          // 57.8MB
    _Float16* wqkvT   = (_Float16*)(ws + off); off += (size_t)HOUT * DIM_ * 2;         // 4.0MB
    _Float16* wprojT  = (_Float16*)(ws + off); off += (size_t)DIM_ * DHDIM * 2;        // 2.7MB
    float*    biasM   = (float*)(ws + off);    off += (size_t)NHEADS * NTOK * NTOK * 4;// 2.8MB
    _Float16* qBuf    = (_Float16*)(ws + off); off += (size_t)BATCH * NHEADS * NTOK * KD * 2;
    _Float16* kBuf    = (_Float16*)(ws + off); off += (size_t)BATCH * NHEADS * NTOK * KD * 2;
    _Float16* vTBuf   = (_Float16*)(ws + off); off += (size_t)BATCH * NHEADS * VD * VSTRIDE * 2;
    _Float16* attnOut = (_Float16*)(ws + off); off += (size_t)TOK * DHDIM * 2;

    k_prep_wqkv <<<(HOUT * DIM_ + 255) / 256, 256, 0, stream>>>(w_qkv, wqkvT);
    k_prep_wproj<<<(DIM_ * DHDIM + 255) / 256, 256, 0, stream>>>(w_proj, wprojT);
    k_prep_bias <<<(NHEADS * NTOK * NTOK + 255) / 256, 256, 0, stream>>>(attn_b, bidx, biasM, offset_size);
    k_ln<<<TOK / 8, 256, 0, stream>>>(x, ln_w, ln_b, xn);
    k_gemm_qkv<<<dim3(HOUT / 128, TOK / 128), 256, 0, stream>>>(xn, wqkvT, b_qkv, qBuf, kBuf, vTBuf);
    k_attn<<<BATCH * NHEADS, 256, 0, stream>>>(qBuf, kBuf, vTBuf, biasM, attnOut);
    k_gemm_proj<<<dim3((DIM_ + 127) / 128, TOK / 128), 256, 0, stream>>>(attnOut, wprojT, b_proj, out);
}
